// GCN_89283780149800
// MI455X (gfx1250) — matrix operand order repaired and checked by reading, not compiled, not run on hardware
//
#include <hip/hip_runtime.h>

typedef __attribute__((ext_vector_type(16))) _Float16 v16h;
typedef __attribute__((ext_vector_type(8)))  _Float16 v8h;
typedef __attribute__((ext_vector_type(8)))  float    v8f;

#define HF 128   // hidden feature dim (fixed by the model)

__device__ __forceinline__ void atomic_add_f32(float* p, float v) {
    // Non-returning agent-scope fp32 add -> global_atomic_add_f32
    __hip_atomic_fetch_add(p, v, __ATOMIC_RELAXED, __HIP_MEMORY_SCOPE_AGENT);
}

// ---------------- degree / normalization ----------------
__global__ void k_deg_init(float* deg, int n) {
    int i = blockIdx.x * blockDim.x + threadIdx.x;
    if (i < n) deg[i] = 1.0f;                 // self-loop contributes 1
}

__global__ void k_deg_edges(const int* __restrict__ dst, float* deg, int e) {
    int i = blockIdx.x * blockDim.x + threadIdx.x;
    if (i < e) atomic_add_f32(&deg[dst[i]], 1.0f);
}

__global__ void k_dinv(const float* __restrict__ deg, float* dinv, int n) {
    int i = blockIdx.x * blockDim.x + threadIdx.x;
    if (i < n) dinv[i] = rsqrtf(deg[i]);      // deg >= 1 always
}

// ---------------- weight convert: Wt[n][k] = (f16) W[k][n], col-padded ----------------
__global__ void k_wt(const float* __restrict__ W, _Float16* __restrict__ Wt,
                     int ncols, int ncolsPad) {
    int i = blockIdx.x * blockDim.x + threadIdx.x;
    if (i >= ncolsPad * HF) return;
    int n = i / HF, k = i % HF;
    Wt[n * HF + k] = (n < ncols) ? (_Float16)W[k * ncols + n] : (_Float16)0.0f;
}

// ---------------- WMMA GEMM ----------------
// C[nrows x ncols] = act(A) @ W, act(a) = tanh(a + abias[k]) if ABIAS else a.
// If OBIAS:  C += bias[col].
// If CINIT:  also writes Cinit[row] = C[row] * dinv[row]^2  (self-loop seed for the
//            scatter-add that follows).
// Launch blockDim.x == 32 * (ncolsPad/16); each block computes a 64-row x ncolsPad
// tile with K = 128 fully LDS-resident.
template <bool ABIAS, bool OBIAS, bool CINIT>
__global__ __launch_bounds__(256) void k_gemm(
    const float* __restrict__ A, const _Float16* __restrict__ Wt,
    const float* __restrict__ abias,
    const float* __restrict__ bias,
    const float* __restrict__ dinv,
    float* __restrict__ C,
    float* __restrict__ Cinit,
    int nrows, int ncols, int ncolsPad) {

    __shared__ __align__(32) _Float16 wlds[HF * HF];   // up to 128x128 f16 (32 KB)
    __shared__ __align__(32) _Float16 alds[64 * HF];   // 64x128 f16 (16 KB)

    const int tid      = threadIdx.x;
    const int nthreads = blockDim.x;
    const int rowbase  = blockIdx.x * 64;

    // Stage transposed weights (K-contiguous) into LDS, 16B chunks
    const int wcnt = (ncolsPad * HF) / 8;
    for (int i = tid; i < wcnt; i += nthreads)
        *(uint4*)&wlds[i * 8] = *(const uint4*)&Wt[i * 8];

    // Stage A tile: 64 x 128 f32 -> (optional bias+tanh) -> f16, packed 8B LDS stores
    for (int i = tid; i < 64 * HF / 4; i += nthreads) {
        int r  = i >> 5;               // 32 float4 per row
        int c4 = (i & 31) * 4;
        int gr = rowbase + r;
        float4 v = make_float4(0.f, 0.f, 0.f, 0.f);
        if (gr < nrows) v = *(const float4*)&A[(size_t)gr * HF + c4];
        if (ABIAS) {
            float4 ab = *(const float4*)&abias[c4];
            v.x = tanhf(v.x + ab.x); v.y = tanhf(v.y + ab.y);
            v.z = tanhf(v.z + ab.z); v.w = tanhf(v.w + ab.w);
        }
        union { _Float16 h[4]; uint2 u; } pk;
        pk.h[0] = (_Float16)v.x; pk.h[1] = (_Float16)v.y;
        pk.h[2] = (_Float16)v.z; pk.h[3] = (_Float16)v.w;
        *(uint2*)&alds[r * HF + c4] = pk.u;
    }
    __syncthreads();

    const int wave = tid >> 5;
    const int lane = tid & 31;
    const int half = lane >> 4;
    const int lm   = lane & 15;
    const int n0   = wave * 16;        // this wave's column tile
    const int col  = n0 + lm;

    // Hoisted output-bias value for this lane's column
    float bval = 0.0f;
    if (OBIAS) { if (col < ncols) bval = bias[col]; }

    // B fragments: lane holds column n0+lm, K = 32*kk + 16*half .. +15 (contiguous)
    v16h bfr[4];
#pragma unroll
    for (int kk = 0; kk < 4; ++kk)
        bfr[kk] = *(const v16h*)&wlds[(n0 + lm) * HF + kk * 32 + half * 16];

    const bool fullTile = (rowbase + 64 <= nrows) && (ncols == ncolsPad);

#pragma unroll
    for (int rt = 0; rt < 4; ++rt) {
        const int r0 = rt * 16;
        v8f acc = {};
#pragma unroll
        for (int kk = 0; kk < 4; ++kk) {
            // A 16x32 f16 layout: lanes 0-15 K=0..15, lanes 16-31 K=16..31 (contiguous, matches B)
            const _Float16* ap = &alds[(r0 + lm) * HF + kk * 32 + half * 16];
            v8h lo = *(const v8h*)ap;
            v8h hi = *(const v8h*)(ap + 8);
            v16h af = __builtin_shufflevector(lo, hi,
                        0, 1, 2, 3, 4, 5, 6, 7, 8, 9, 10, 11, 12, 13, 14, 15);
            acc = __builtin_amdgcn_wmma_f32_16x16x32_f16(
                      false, af, false, bfr[kk], (short)0, acc, false, false);
        }
        const int row0 = rowbase + r0 + half * 8;   // first of this lane's 8 rows
        float* cp = &C[(size_t)row0 * ncols + col];
        float* ip = nullptr;
        if (CINIT) ip = &Cinit[(size_t)row0 * ncols + col];
        if (fullTile) {
#pragma unroll
            for (int v = 0; v < 8; ++v) {
                float x = acc[v] + bval;
                cp[0] = x;
                cp += ncols;
                if (CINIT) {
                    float di = dinv[row0 + v];   // uniform across the 16-lane half
                    ip[0] = x * (di * di);
                    ip += ncols;
                }
            }
        } else {
#pragma unroll
            for (int v = 0; v < 8; ++v) {
                int row = row0 + v;
                if (row < nrows && col < ncols) {
                    float x = acc[v] + bval;
                    cp[0] = x;
                    if (CINIT) {
                        float di = dinv[row];
                        ip[0] = x * (di * di);
                    }
                }
                cp += ncols;
                if (CINIT) ip += ncols;
            }
        }
    }
}

// ---------------- edge scatter: out[dst] += h[src] * dinv[src]*dinv[dst] ----------------
// One wave handles one edge (32 lanes x 4 floats = 128 features).
__global__ void k_agg(const float* __restrict__ h, const int* __restrict__ src,
                      const int* __restrict__ dst, const float* __restrict__ dinv,
                      float* out, int e) {
    long long i = (long long)blockIdx.x * blockDim.x + threadIdx.x;
    long long total = (long long)e * 32;
    if (i >= total) return;                 // total % 32 == 0: whole waves exit together
    int eid = (int)(i >> 5);
    eid = __builtin_amdgcn_readfirstlane(eid);   // wave-uniform -> scalar loads below
    int g = ((int)i & 31) * 4;
    int s = src[eid];
    int d = dst[eid];
    float nm = dinv[s] * dinv[d];
    float4 v = *(const float4*)&h[(size_t)s * HF + g];   // coalesced 512B/edge gather
    float* o = &out[(size_t)d * HF + g];
    atomic_add_f32(o + 0, v.x * nm);
    atomic_add_f32(o + 1, v.y * nm);
    atomic_add_f32(o + 2, v.z * nm);
    atomic_add_f32(o + 3, v.w * nm);
}

// ---------------- bias + tanh (in place, layer-3 output only) ----------------
__global__ void k_bias_tanh(float* h, const float* __restrict__ b, int n) {
    long long i = (long long)blockIdx.x * blockDim.x + threadIdx.x;
    if (i >= (long long)n * HF) return;
    int f = (int)(i & (HF - 1));
    h[i] = tanhf(h[i] + b[f]);
}

extern "C" void kernel_launch(void* const* d_in, const int* in_sizes, int n_in,
                              void* d_out, int out_size, void* d_ws, size_t ws_size,
                              hipStream_t stream) {
    const float* x  = (const float*)d_in[0];
    const int*   ei = (const int*)d_in[1];
    const float* W1 = (const float*)d_in[2]; const float* b1 = (const float*)d_in[3];
    const float* W2 = (const float*)d_in[4]; const float* b2 = (const float*)d_in[5];
    const float* W3 = (const float*)d_in[6]; const float* b3 = (const float*)d_in[7];
    const float* Wc = (const float*)d_in[8]; const float* bc = (const float*)d_in[9];

    const int N = in_sizes[0] / HF;        // 100000
    const int E = in_sizes[1] / 2;         // 3200000
    const int C = in_sizes[9];             // 40
    const int CPad = ((C + 15) / 16) * 16; // 48

    const int* src = ei;
    const int* dst = ei + E;

    float* out  = (float*)d_out;                 // [N, C]
    float* hout = out + (size_t)N * C;           // [N, HF] (second tuple output)

    // workspace layout (floats)
    float* ws   = (float*)d_ws;
    float* dinv = ws;                            // N
    float* deg  = ws + 131072;                   // N
    float* B0   = ws + 262144;                   // N*HF (linear / GEMM out)
    float* B1   = B0 + (size_t)N * HF;           // N*HF (aggregate buffer)
    _Float16* w1t = (_Float16*)(B1 + (size_t)N * HF);  // 16B-aligned by construction
    _Float16* w2t = w1t + HF * HF;
    _Float16* w3t = w2t + HF * HF;
    _Float16* wct = w3t + HF * HF;               // 48*128 used

    const int T = 256;
    const int nBlkN   = (N + T - 1) / T;
    const int nBlkE   = (E + T - 1) / T;
    const int nBlkW   = (HF * HF + T - 1) / T;
    const int nBlkWc  = (CPad * HF + T - 1) / T;
    const int gemmBlk = (N + 63) / 64;
    const long long aggItems = (long long)E * 32;
    const int aggBlk  = (int)((aggItems + T - 1) / T);
    const long long nh = (long long)N * HF;
    const int btBlk   = (int)((nh + T - 1) / T);

    // normalization
    k_deg_init <<<nBlkN, T, 0, stream>>>(deg, N);
    k_deg_edges<<<nBlkE, T, 0, stream>>>(dst, deg, E);
    k_dinv     <<<nBlkN, T, 0, stream>>>(deg, dinv, N);

    // weights -> f16 transposed
    k_wt<<<nBlkW,  T, 0, stream>>>(W1, w1t, HF, HF);
    k_wt<<<nBlkW,  T, 0, stream>>>(W2, w2t, HF, HF);
    k_wt<<<nBlkW,  T, 0, stream>>>(W3, w3t, HF, HF);
    k_wt<<<nBlkWc, T, 0, stream>>>(Wc, wct, C, CPad);

    // layer 1: lin1 = x@W1 -> B0, hout seeded with self-loop term, scatter into hout
    k_gemm<false, false, true><<<gemmBlk, 256, 0, stream>>>(
        x, w1t, nullptr, nullptr, dinv, B0, hout, N, HF, HF);
    k_agg<<<aggBlk, T, 0, stream>>>(B0, src, dst, dinv, hout, E);
    // hout now holds raw agg1 (pre bias/tanh); activation fused into layer-2 staging

    // layer 2: lin2 = tanh(agg1+b1)@W2 -> B0, B1 seeded, scatter into B1
    k_gemm<true, false, true><<<gemmBlk, 256, 0, stream>>>(
        hout, w2t, b1, nullptr, dinv, B0, B1, N, HF, HF);
    k_agg<<<aggBlk, T, 0, stream>>>(B0, src, dst, dinv, B1, E);

    // layer 3: lin3 = tanh(agg2+b2)@W3 -> B0, hout seeded, scatter into hout
    k_gemm<true, false, true><<<gemmBlk, 256, 0, stream>>>(
        B1, w3t, b2, nullptr, dinv, B0, hout, N, HF, HF);
    k_agg<<<aggBlk, T, 0, stream>>>(B0, src, dst, dinv, hout, E);
    // materialize final h3 = tanh(agg3 + b3) (tuple output)
    k_bias_tanh<<<btBlk, T, 0, stream>>>(hout, b3, N);

    // classifier: out = h3 @ Wc + bc  (3 waves = 3 col tiles of 16, 48-padded)
    k_gemm<false, true, false><<<gemmBlk, 32 * (CPad / 16), 0, stream>>>(
        hout, wct, nullptr, bc, nullptr, out, nullptr, N, C, CPad);
}
